// SelfAttention_4363686772737
// MI455X (gfx1250) — matrix operand order repaired and checked by reading, not compiled, not run on hardware
//
#include <hip/hip_runtime.h>

// ---------------------------------------------------------------------------
// Problem constants (from reference): B=2, S=2048, D=2048, H=16, DQ=128
// ---------------------------------------------------------------------------
#define BB   2
#define SS   2048
#define DD   2048
#define HH   16
#define DQQ  128
#define NQKV (3 * HH * DQQ)      // 6144
#define MROW (BB * SS)           // 4096
#define HD   (HH * DQQ)          // 2048

typedef __attribute__((ext_vector_type(16))) __bf16   bf16x16;
typedef __attribute__((ext_vector_type(8)))  float    f32x8;
typedef __attribute__((ext_vector_type(4)))  unsigned u32x4;

union Frag { bf16x16 v; u32x4 u[2]; };

// ---- bf16 <-> f32 helpers (manual, RNE) -----------------------------------
static __device__ inline unsigned short f2bf(float f) {
    unsigned u = __float_as_uint(f);
    u += 0x7fffu + ((u >> 16) & 1u);
    return (unsigned short)(u >> 16);
}
static __device__ inline float bf2f(unsigned short h) {
    return __uint_as_float(((unsigned)h) << 16);
}

// ---- WMMA fragment loaders (CDNA5 16x16x32 bf16 layouts) -------------------
// A (16x32, MxK): lane<16: elems = K{0..7,16..23}; lane>=16: K{8..15,24..31}
static __device__ inline Frag load_a(const unsigned short* p, int ld, int lane) {
    const int row = lane & 15;
    const int hi  = lane >> 4;
    const unsigned short* q = p + row * ld + hi * 8;
    Frag f;
    f.u[0] = *reinterpret_cast<const u32x4*>(q);
    f.u[1] = *reinterpret_cast<const u32x4*>(q + 16);
    return f;
}
// B (32x16, KxN) loaded from row-major B^T [N][K]: lane<16: K{0..7,16..23}; lane>=16: K{8..15,24..31}
static __device__ inline Frag load_b(const unsigned short* p, int ld, int lane) {
    const int row = lane & 15;
    const int hi  = lane >> 4;
    const unsigned short* q = p + row * ld + hi * 8;
    Frag f;
    f.u[0] = *reinterpret_cast<const u32x4*>(q);
    f.u[1] = *reinterpret_cast<const u32x4*>(q + 16);
    return f;
}

static __device__ inline f32x8 wmma_bf16(const Frag& a, const Frag& b, f32x8 c) {
    return __builtin_amdgcn_wmma_f32_16x16x32_bf16(
        /*neg_a=*/false, a.v, /*neg_b=*/false, b.v,
        /*c_mod=*/(short)0, c, /*reuse_a=*/false, /*reuse_b=*/false);
}

// ---- CDNA5 async global->LDS copy (16B per lane), tracked by ASYNCcnt ------
// LDS dest is the lane's VGPR offset plus the instruction offset ; LDS addr = generic addr[31:0]
static __device__ inline void async_copy16(unsigned lds_off, const unsigned short* gptr) {
    asm volatile("global_load_async_to_lds_b128 %0, %1, off"
                 :: "v"(lds_off), "v"((unsigned long long)gptr)
                 : "memory");
}
static __device__ inline void wait_async0() {
    asm volatile("s_wait_asynccnt 0" ::: "memory");
}

// ---------------------------------------------------------------------------
// K0: f32 -> bf16 elementwise
// ---------------------------------------------------------------------------
__global__ void conv_kernel(const float* __restrict__ src,
                            unsigned short* __restrict__ dst, int n) {
    int idx = blockIdx.x * blockDim.x + threadIdx.x;
    if (idx < n) dst[idx] = f2bf(src[idx]);
}

// K0b: f32 [R][N] -> bf16 transposed [N][R]
__global__ void convT_kernel(const float* __restrict__ src,
                             unsigned short* __restrict__ dst, int R, int N) {
    int idx = blockIdx.x * blockDim.x + threadIdx.x;
    if (idx < R * N) {
        int r = idx / N;
        int c = idx - r * N;
        dst[(size_t)c * R + r] = f2bf(src[idx]);
    }
}

// ---------------------------------------------------------------------------
// GEMM: C[M,N] = A[M,K](bf16) * Bt[N,K]^T(bf16) + bias
// block = 256 threads (8 waves), block tile 128x128, wave tile 32x64
// ---------------------------------------------------------------------------
template <bool OUT_BF16>
__global__ __launch_bounds__(256) void gemm_kernel(
    const unsigned short* __restrict__ A, const unsigned short* __restrict__ Bt,
    const float* __restrict__ bias, void* __restrict__ Cout,
    int M, int N, int K) {
    const int lane = threadIdx.x & 31;
    const int wave = threadIdx.x >> 5;
    const int wm   = wave >> 1;           // 0..3
    const int wn   = wave & 1;            // 0..1
    const int m0   = blockIdx.y * 128 + wm * 32;
    const int n0   = blockIdx.x * 128 + wn * 64;

    f32x8 acc[2][4] = {};

    for (int k = 0; k < K; k += 32) {
        if (k + 128 < K) {   // uniform branch; prefetch next tiles (global_prefetch_b8)
            __builtin_prefetch(A + (size_t)(m0 + (lane & 15)) * K + k + 128, 0, 0);
            __builtin_prefetch(Bt + (size_t)(n0 + lane) * K + k + 128, 0, 0);
        }
        Frag a0 = load_a(A + (size_t)m0 * K + k, K, lane);
        Frag a1 = load_a(A + (size_t)(m0 + 16) * K + k, K, lane);
        Frag b[4];
#pragma unroll
        for (int j = 0; j < 4; ++j)
            b[j] = load_b(Bt + (size_t)(n0 + j * 16) * K + k, K, lane);
#pragma unroll
        for (int j = 0; j < 4; ++j) {
            acc[0][j] = wmma_bf16(a0, b[j], acc[0][j]);
            acc[1][j] = wmma_bf16(a1, b[j], acc[1][j]);
        }
    }

    const int hi  = lane >> 4;
    const int c16 = lane & 15;
#pragma unroll
    for (int i = 0; i < 2; ++i)
#pragma unroll
        for (int j = 0; j < 4; ++j)
#pragma unroll
            for (int r = 0; r < 8; ++r) {
                int rowg = m0 + i * 16 + r + hi * 8;
                int colg = n0 + j * 16 + c16;
                float v = acc[i][j][r] + bias[colg];
                if (OUT_BF16)
                    ((unsigned short*)Cout)[(size_t)rowg * N + colg] = f2bf(v);
                else
                    ((float*)Cout)[(size_t)rowg * N + colg] = v;
            }
}

// ---------------------------------------------------------------------------
// RoPE + scatter: qkv bf16 [B*S, 3*H*DQ] -> q,k [B,H,S,DQ] bf16, v^T [B,H,DQ,S]
// one thread per (b,s,h,j) with j in [0,64): handles pair (j, j+64)
// ---------------------------------------------------------------------------
__global__ void rope_kernel(const unsigned short* __restrict__ qkv,
                            const int* __restrict__ pos,
                            unsigned short* __restrict__ qb,
                            unsigned short* __restrict__ kb,
                            unsigned short* __restrict__ vt) {
    int idx = blockIdx.x * blockDim.x + threadIdx.x;   // [0, B*S*H*64)
    int j = idx & 63;
    int h = (idx >> 6) & (HH - 1);
    int s = (idx >> 10) & (SS - 1);
    int b = idx >> 21;

    size_t base = (size_t)(b * SS + s) * NQKV;
    float q1 = bf2f(qkv[base + (0 * HH + h) * DQQ + j]);
    float q2 = bf2f(qkv[base + (0 * HH + h) * DQQ + j + 64]);
    float k1 = bf2f(qkv[base + (1 * HH + h) * DQQ + j]);
    float k2 = bf2f(qkv[base + (1 * HH + h) * DQQ + j + 64]);
    float v1 = bf2f(qkv[base + (2 * HH + h) * DQQ + j]);
    float v2 = bf2f(qkv[base + (2 * HH + h) * DQQ + j + 64]);

    float p = (float)pos[s];
    float inv_freq = powf(10000.0f, -(float)j * (1.0f / 64.0f));
    float ang = p * inv_freq;
    float sn, cs;
    sincosf(ang, &sn, &cs);

    size_t qk_base = ((size_t)(b * HH + h) * SS + s) * DQQ;
    qb[qk_base + j]      = f2bf(q1 * cs - q2 * sn);
    qb[qk_base + j + 64] = f2bf(q1 * sn + q2 * cs);
    kb[qk_base + j]      = f2bf(k1 * cs - k2 * sn);
    kb[qk_base + j + 64] = f2bf(k1 * sn + k2 * cs);

    size_t vt_base = (size_t)(b * HH + h) * DQQ;
    vt[(vt_base + j)      * SS + s] = f2bf(v1);
    vt[(vt_base + j + 64) * SS + s] = f2bf(v2);
}

// ---------------------------------------------------------------------------
// Causal flash attention: grid(S/64, B*H), block 128 (4 waves, 16 q-rows each)
// K/V tiles staged once per block via async global->LDS; WMMAs fed from LDS.
// ---------------------------------------------------------------------------
__global__ __launch_bounds__(128) void attn_kernel(
    const unsigned short* __restrict__ Q, const unsigned short* __restrict__ Km,
    const unsigned short* __restrict__ Vt, unsigned short* __restrict__ Oc) {
    const int lane = threadIdx.x & 31;
    const int wave = threadIdx.x >> 5;
    const int bh = blockIdx.y;
    const int b = bh >> 4, h = bh & 15;
    const int q0 = blockIdx.x * 64 + wave * 16;

    const unsigned short* qh = Q  + (size_t)bh * SS * DQQ;
    const unsigned short* kh = Km + (size_t)bh * SS * DQQ;
    const unsigned short* vh = Vt + (size_t)bh * DQQ * SS;

    Frag qf[4];
#pragma unroll
    for (int c = 0; c < 4; ++c)
        qf[c] = load_a(qh + (size_t)q0 * DQQ + c * 32, DQQ, lane);

    f32x8 acc[8] = {};
    float mrun[8], lrun[8];
#pragma unroll
    for (int r = 0; r < 8; ++r) { mrun[r] = -3.0e38f; lrun[r] = 0.0f; }

    // LDS: K tile [32 kv][128 dq] (8KB), V^T tile [128 dq][32 kv] (8KB), P (4KB)
    __shared__ alignas(16) unsigned short ktile[32 * 128];
    __shared__ alignas(16) unsigned short vtile[128 * 32];
    __shared__ alignas(16) unsigned short pbuf[4][16 * 32];

    const unsigned ktile_base = (unsigned)(unsigned long long)&ktile[0];
    const unsigned vtile_base = (unsigned)(unsigned long long)&vtile[0];

    const int hi  = lane >> 4;
    const int c16 = lane & 15;
    const int kv_end = blockIdx.x * 64 + 64;
    const float scale = 0.08838834764831845f;   // 1/sqrt(128)

    for (int kv = 0; kv < kv_end; kv += 32) {
        // ---- async stage of K/V tiles: 512 chunks of 16B each, 128 threads ----
#pragma unroll
        for (int i = 0; i < 4; ++i) {
            int c = (int)threadIdx.x + i * 128;
            int krow = c >> 4, kcol = c & 15;             // K tile: 16 chunks/row
            async_copy16(ktile_base + c * 16,
                         kh + (size_t)(kv + krow) * DQQ + kcol * 8);
            int vrow = c >> 2, vcol = c & 3;              // V tile: 4 chunks/row
            async_copy16(vtile_base + c * 16,
                         vh + (size_t)vrow * SS + kv + vcol * 8);
        }
        wait_async0();
        __syncthreads();

        // ---- scores: S = Q K^T over DQ=128 (4 WMMAs per 16-col tile) ----
        f32x8 sc0 = {}, sc1 = {};
#pragma unroll
        for (int c = 0; c < 4; ++c) {
            Frag k0 = load_b(ktile + 0 * 128 + c * 32, 128, lane);
            Frag k1 = load_b(ktile + 16 * 128 + c * 32, 128, lane);
            sc0 = wmma_bf16(qf[c], k0, sc0);
            sc1 = wmma_bf16(qf[c], k1, sc1);
        }
        // ---- causal mask + online softmax (C-layout row stats) ----
#pragma unroll
        for (int r = 0; r < 8; ++r) {
            int qr = q0 + r + hi * 8;
            float s0 = sc0[r] * scale;
            float s1 = sc1[r] * scale;
            s0 = ((kv + c16)      <= qr) ? s0 : -3.0e38f;
            s1 = ((kv + 16 + c16) <= qr) ? s1 : -3.0e38f;
            float mx = fmaxf(s0, s1);
            mx = fmaxf(mx, __shfl_xor(mx, 1, 32));
            mx = fmaxf(mx, __shfl_xor(mx, 2, 32));
            mx = fmaxf(mx, __shfl_xor(mx, 4, 32));
            mx = fmaxf(mx, __shfl_xor(mx, 8, 32));
            float mnew  = fmaxf(mrun[r], mx);
            float alpha = __expf(mrun[r] - mnew);
            mrun[r] = mnew;
            float p0 = __expf(s0 - mnew);
            float p1 = __expf(s1 - mnew);
            float rs = p0 + p1;
            rs += __shfl_xor(rs, 1, 32);
            rs += __shfl_xor(rs, 2, 32);
            rs += __shfl_xor(rs, 4, 32);
            rs += __shfl_xor(rs, 8, 32);
            lrun[r] = lrun[r] * alpha + rs;
#pragma unroll
            for (int d = 0; d < 8; ++d) acc[d][r] *= alpha;
            int rl = r + hi * 8;
            pbuf[wave][rl * 32 + c16]      = f2bf(p0);
            pbuf[wave][rl * 32 + 16 + c16] = f2bf(p1);
        }
        __syncthreads();
        // ---- O += P V : A = P(16x32) from LDS, B = V^T rows from LDS ----
        Frag pf = load_a(&pbuf[wave][0], 32, lane);
#pragma unroll
        for (int d = 0; d < 8; ++d) {
            Frag vf = load_b(vtile + (d * 16) * 32, 32, lane);
            acc[d] = wmma_bf16(pf, vf, acc[d]);
        }
        __syncthreads();   // tiles + pbuf dead before next iteration's writes
    }

#pragma unroll
    for (int r = 0; r < 8; ++r) {
        float inv = 1.0f / lrun[r];
        int rowg = b * SS + q0 + r + hi * 8;
#pragma unroll
        for (int d = 0; d < 8; ++d)
            Oc[(size_t)rowg * HD + h * DQQ + d * 16 + c16] = f2bf(acc[d][r] * inv);
    }
}

// ---------------------------------------------------------------------------
// Host launch
// ---------------------------------------------------------------------------
extern "C" void kernel_launch(void* const* d_in, const int* in_sizes, int n_in,
                              void* d_out, int out_size, void* d_ws, size_t ws_size,
                              hipStream_t stream) {
    const float* x        = (const float*)d_in[0];
    const int*   rope_pos = (const int*)d_in[1];
    const float* W_qkv    = (const float*)d_in[2];
    const float* b_qkv    = (const float*)d_in[3];
    const float* W_out    = (const float*)d_in[4];
    const float* b_out    = (const float*)d_in[5];
    float* out = (float*)d_out;

    // workspace layout (ushort elements)
    unsigned short* ws = (unsigned short*)d_ws;
    unsigned short* xb     = ws;                               //  4096*2048
    unsigned short* wqkvT  = xb    + (size_t)MROW * DD;        //  6144*2048
    unsigned short* woutT  = wqkvT + (size_t)NQKV * DD;        //  2048*2048
    unsigned short* qkv    = woutT + (size_t)DD * HD;          //  4096*6144
    unsigned short* qb     = qkv   + (size_t)MROW * NQKV;      //  2*16*2048*128
    unsigned short* kb     = qb    + (size_t)BB * HH * SS * DQQ;
    unsigned short* vt     = kb    + (size_t)BB * HH * SS * DQQ;
    unsigned short* attn   = vt    + (size_t)BB * HH * SS * DQQ;

    // 1) convert x -> bf16
    conv_kernel<<<(MROW * DD) / 256, 256, 0, stream>>>(x, xb, MROW * DD);
    // 2) W_qkv [D, 3HD] -> bf16 transposed [3HD, D]
    convT_kernel<<<(DD * NQKV) / 256, 256, 0, stream>>>(W_qkv, wqkvT, DD, NQKV);
    // 3) W_out [HD, D] -> bf16 transposed [D, HD]
    convT_kernel<<<(HD * DD) / 256, 256, 0, stream>>>(W_out, woutT, HD, DD);
    // 4) QKV GEMM: [4096,6144] = xb @ wqkvT^T + b_qkv  (bf16 out)
    gemm_kernel<true><<<dim3(NQKV / 128, MROW / 128), 256, 0, stream>>>(
        xb, wqkvT, b_qkv, (void*)qkv, MROW, NQKV, DD);
    // 5) RoPE + scatter to q,k,[v^T]
    rope_kernel<<<(BB * SS * HH * 64) / 256, 256, 0, stream>>>(qkv, rope_pos, qb, kb, vt);
    // 6) causal flash attention
    attn_kernel<<<dim3(SS / 64, BB * HH), 128, 0, stream>>>(qb, kb, vt, attn);
    // 7) output projection: out = attn @ W_out + b_out  (fp32 out)
    gemm_kernel<false><<<dim3(DD / 128, MROW / 128), 256, 0, stream>>>(
        attn, woutT, b_out, (void*)out, MROW, DD, HD);
}
